// QueryCrossAttention_60610578481566
// MI455X (gfx1250) — matrix operand order repaired and checked by reading, not compiled, not run on hardware
//
#include <hip/hip_runtime.h>
#include <hip/hip_bf16.h>

typedef __attribute__((ext_vector_type(16))) __bf16 v16bf;
typedef __attribute__((ext_vector_type(8)))  float  v8f;
typedef __attribute__((ext_vector_type(4)))  float  v4f;
typedef __attribute__((ext_vector_type(4)))  unsigned int v4u;
typedef __attribute__((ext_vector_type(2)))  unsigned int v2u;
typedef unsigned short u16;

// A/B fragment: 16 bf16 = two 16-byte chunks (see CDNA5 ISA 7.12.2)
union Frag { v4u u[2]; v16bf b; };

__device__ __forceinline__ u16 f2bf(float f) {  // round-to-nearest-even f32->bf16
  unsigned int x = __builtin_bit_cast(unsigned int, f);
  return (u16)((x + 0x7FFFu + ((x >> 16) & 1u)) >> 16);
}

#define WMMA_BF16(A, B, C) \
  __builtin_amdgcn_wmma_f32_16x16x32_bf16(false, (A), false, (B), (short)0, (C), false, false)

// ---------------------------------------------------------------------------
// DPP butterfly all-reduce within each 16-lane group (VALU-only; co-executes
// with the XDL/WMMA pipe, unlike ds_bpermute).
// ctrls: quad_perm xor1 = 0xB1, quad_perm xor2 = 0x4E,
//        row_half_mirror = 0x141 (xor7 in 8), row_mirror = 0x140 (xor15 in 16)
// ---------------------------------------------------------------------------
template <int CTRL>
__device__ __forceinline__ float dpp_f32(float x) {
  return __builtin_bit_cast(float,
      __builtin_amdgcn_update_dpp(0, __builtin_bit_cast(int, x), CTRL, 0xF, 0xF, true));
}
__device__ __forceinline__ float grp16_max(float x) {
  x = fmaxf(x, dpp_f32<0xB1>(x));
  x = fmaxf(x, dpp_f32<0x4E>(x));
  x = fmaxf(x, dpp_f32<0x141>(x));
  x = fmaxf(x, dpp_f32<0x140>(x));
  return x;
}
__device__ __forceinline__ float grp16_sum(float x) {
  x += dpp_f32<0xB1>(x);
  x += dpp_f32<0x4E>(x);
  x += dpp_f32<0x141>(x);
  x += dpp_f32<0x140>(x);
  return x;
}

// ---------------------------------------------------------------------------
// CDNA5 async global->LDS copy (ASYNCcnt path), with synchronous fallback.
// ---------------------------------------------------------------------------
#if defined(__has_builtin)
#if __has_builtin(__builtin_amdgcn_global_load_async_to_lds_b128) && \
    __has_builtin(__builtin_amdgcn_s_wait_asynccnt)
#define USE_ASYNC_LDS 1
#endif
#endif
#ifndef USE_ASYNC_LDS
#define USE_ASYNC_LDS 0
#endif

typedef int v4i __attribute__((vector_size(16)));
typedef __attribute__((address_space(1))) v4i g_v4i;   // global-memory v4i
typedef __attribute__((address_space(3))) v4i l_v4i;   // LDS v4i

__device__ __forceinline__ void cp16(u16* lds_dst, const u16* gsrc) {
#if USE_ASYNC_LDS
  __builtin_amdgcn_global_load_async_to_lds_b128(
      (g_v4i*)gsrc, (l_v4i*)lds_dst, 0, 0);
#else
  *(v4u*)lds_dst = *(const v4u*)gsrc;
#endif
}

template <short N>
__device__ __forceinline__ void cp_waitN() {  // allow N async ops outstanding
#if USE_ASYNC_LDS
  __builtin_amdgcn_s_wait_asynccnt(N);
#endif
}

// ---------------------------------------------------------------------------
// Weight transpose + bf16 cast: W(K=1024, N=1024) f32  ->  Wt(N, K) bf16
// ---------------------------------------------------------------------------
__global__ __launch_bounds__(256) void wtrans_k(const float* __restrict__ W,
                                                u16* __restrict__ Wt) {
  __shared__ float t[32 * 33];
  const int k0 = (blockIdx.x & 31) * 32, n0 = (blockIdx.x >> 5) * 32;
  #pragma unroll
  for (int e = 0; e < 4; e++) {
    const int idx = threadIdx.x + 256 * e;
    const int rr = idx >> 5, cc = idx & 31;
    t[rr * 33 + cc] = W[(long)(k0 + rr) * 1024 + n0 + cc];
  }
  __syncthreads();
  #pragma unroll
  for (int e = 0; e < 4; e++) {
    const int idx = threadIdx.x + 256 * e;
    const int nr = idx >> 5, kc = idx & 31;
    Wt[(long)(n0 + nr) * 1024 + k0 + kc] = f2bf(t[kc * 33 + nr]);
  }
}

// ---------------------------------------------------------------------------
// LayerNorm f32 -> bf16, one wave32 per 1024-element row.
// Output goes into the concatenated kv_input buffer: (b*2560 + rowOff + r)*1024
// ---------------------------------------------------------------------------
__global__ __launch_bounds__(256) void ln_bf16_k(const float* __restrict__ x,
                                                 const float* __restrict__ g,
                                                 const float* __restrict__ be,
                                                 u16* __restrict__ out,
                                                 int rpb, int rowOff) {
  const int row = blockIdx.x * 8 + (threadIdx.x >> 5);
  const int lane = threadIdx.x & 31;
  const int b = row / rpb, r = row % rpb;
  const float* xp = x + (long)row * 1024;
  u16* op = out + ((long)b * 2560 + rowOff + r) * 1024;
  v4f v[8];
  float s = 0.f, s2 = 0.f;
  #pragma unroll
  for (int i = 0; i < 8; i++) {
    v[i] = *(const v4f*)(xp + (lane + 32 * i) * 4);
    #pragma unroll
    for (int c = 0; c < 4; c++) { s += v[i][c]; s2 += v[i][c] * v[i][c]; }
  }
  #pragma unroll
  for (int off = 1; off < 32; off <<= 1) {
    s  += __shfl_xor(s,  off, 32);
    s2 += __shfl_xor(s2, off, 32);
  }
  const float mu  = s  * (1.0f / 1024.0f);
  const float var = s2 * (1.0f / 1024.0f) - mu * mu;
  const float rs  = rsqrtf(var + 1e-5f);
  #pragma unroll
  for (int i = 0; i < 8; i++) {
    const int idx = (lane + 32 * i) * 4;
    v4f gg = *(const v4f*)(g + idx);
    v4f bb = *(const v4f*)(be + idx);
    u16 o[4];
    #pragma unroll
    for (int c = 0; c < 4; c++) o[c] = f2bf((v[i][c] - mu) * rs * gg[c] + bb[c]);
    v2u pk;
    pk[0] = (unsigned)o[0] | ((unsigned)o[1] << 16);
    pk[1] = (unsigned)o[2] | ((unsigned)o[3] << 16);
    *(v2u*)(op + idx) = pk;
  }
}

// ---------------------------------------------------------------------------
// bf16 GEMM: C(M,N) = A(M,K) @ Bt(N,K)^T  (Bt is k-contiguous = W transposed)
// Block tile 128x128, 8 waves, wave tile 32x64, K-step 32.
// 3-stage async pipeline: 3 LDS buffers, one tile kept in flight during
// compute via partial s_wait_asynccnt (4 = one 4-op tile outstanding).
// OUT_MODE 0: bf16 (acc+bias)*scale, row-major
// OUT_MODE 1: bf16 acc+bias, transposed per batch: out[(b*N+n)*ntot + j]
// OUT_MODE 2: f32 acc+bias+residual
// ---------------------------------------------------------------------------
template <int OUT_MODE>
__global__ __launch_bounds__(256) void gemm_bf16_k(
    const u16* __restrict__ A, const u16* __restrict__ Bt,
    const float* __restrict__ bias, const float* __restrict__ resid,
    u16* __restrict__ outB, float* __restrict__ outF,
    int M, int N, int K, int rpb, long aBatch, int aRowOff,
    float scale, int ntot) {
  __shared__ u16 As[3][128 * 40];   // stride 40: 16B-aligned rows, conflict-free
  __shared__ u16 Bs[3][128 * 40];
  const int tid = threadIdx.x, lane = tid & 31, wave = tid >> 5;
  const int half = lane >> 4, l15 = lane & 15;
  const int mBase = blockIdx.y * 128, nBase = blockIdx.x * 128;
  const int wm = (wave >> 1) * 32, wn = (wave & 1) * 64;

  const int lrow = tid >> 1, lk = (tid & 1) * 16;
  const int am = mBase + lrow;
  const int ab = am / rpb, ar = am % rpb;
  const u16* aRow = A + (long)ab * aBatch + (long)(aRowOff + ar) * K;
  const u16* bRow = Bt + (long)(nBase + lrow) * K;
  const int ldsOff = lrow * 40 + lk;

  auto issueTile = [&](int buf, int kb) {   // 4 async b128 ops per thread
    cp16(&As[buf][ldsOff],     aRow + kb + lk);
    cp16(&As[buf][ldsOff + 8], aRow + kb + lk + 8);
    cp16(&Bs[buf][ldsOff],     bRow + kb + lk);
    cp16(&Bs[buf][ldsOff + 8], bRow + kb + lk + 8);
  };

  v8f acc[2][4];
  #pragma unroll
  for (int i = 0; i < 2; i++)
    #pragma unroll
    for (int j = 0; j < 4; j++) acc[i][j] = (v8f)0.0f;

  issueTile(0, 0);                     // prologue: tiles 0 and 1 in flight
  if (32 < K) issueTile(1, 32);
  int cur = 0;
  for (int kb = 0; kb < K; kb += 32) {
    if (kb + 32 < K) cp_waitN<4>();    // tile `cur` done; newest stays in flight
    else             cp_waitN<0>();    // drain on last iteration
    __syncthreads();
    if (kb + 64 < K) {
      int nxt = cur + 2; if (nxt >= 3) nxt -= 3;
      issueTile(nxt, kb + 64);         // overlap tile i+2 with compute of tile i
    }

    Frag af[2], bf[4];
    #pragma unroll
    for (int mi = 0; mi < 2; mi++) {
      const u16* p = &As[cur][(wm + mi * 16 + l15) * 40];
      af[mi].u[0] = *(const v4u*)(p + 16 * half);
      af[mi].u[1] = *(const v4u*)(p + 16 * half + 8);
    }
    #pragma unroll
    for (int ni = 0; ni < 4; ni++) {
      const u16* p = &Bs[cur][(wn + ni * 16 + l15) * 40 + 16 * half];
      bf[ni].u[0] = *(const v4u*)p;
      bf[ni].u[1] = *(const v4u*)(p + 8);
    }
    #pragma unroll
    for (int mi = 0; mi < 2; mi++)
      #pragma unroll
      for (int ni = 0; ni < 4; ni++)
        acc[mi][ni] = WMMA_BF16(af[mi].b, bf[ni].b, acc[mi][ni]);
    cur++; if (cur == 3) cur = 0;
  }

  #pragma unroll
  for (int mi = 0; mi < 2; mi++) {
    #pragma unroll
    for (int ni = 0; ni < 4; ni++) {
      const int col = nBase + wn + ni * 16 + l15;
      const float bv = bias[col];
      const int rowB = mBase + wm + mi * 16 + 8 * half;   // C layout: M = r + 8*half
      #pragma unroll
      for (int r = 0; r < 8; r++) {
        const long orow = rowB + r;
        const float v = acc[mi][ni][r] + bv;
        if constexpr (OUT_MODE == 0) {
          outB[orow * (long)N + col] = f2bf(v * scale);
        } else if constexpr (OUT_MODE == 1) {
          const long bb = orow / ntot, j = orow % ntot;
          outB[(bb * N + col) * (long)ntot + j] = f2bf(v);
        } else {
          const long idx = orow * (long)N + col;
          outF[idx] = v + resid[idx];
        }
      }
    }
  }
}

// ---------------------------------------------------------------------------
// Flash attention: block = (b, h, 128-query tile); 8 waves x 16 rows.
// Double-buffered async K/V staging; DPP softmax reductions.
// qp: (4*512, 1024) bf16, scale pre-applied; kp: (4*2560, 1024) bf16;
// vt: per (b, d=h*64+dd) rows of length 2560 (k-contiguous for P@V B-frags);
// ao: (4*512, 1024) bf16 merged heads.
// ---------------------------------------------------------------------------
__global__ __launch_bounds__(256) void attn_k(const u16* __restrict__ qp,
                                              const u16* __restrict__ kp,
                                              const u16* __restrict__ vt,
                                              u16* __restrict__ ao) {
  constexpr int D = 1024, NTOT = 2560, NQ = 512;
  __shared__ u16 Ks[2][64 * 72];    // stride 72: 16B-aligned, conflict-free
  __shared__ u16 Vs[2][64 * 72];
  __shared__ u16 Ps[8 * 16 * 72];   // wave-private P slabs
  const int tid = threadIdx.x, lane = tid & 31, wave = tid >> 5;
  const int half = lane >> 4, l15 = lane & 15;
  const int mt = blockIdx.x & 3, bh = blockIdx.x >> 2;
  const int h = bh & 15, b = bh >> 4;

  // Q fragments stay in registers for the whole kv sweep
  const long qrow = (long)b * NQ + mt * 128 + wave * 16 + l15;
  const u16* qptr = qp + qrow * D + h * 64;
  Frag qf[2];
  #pragma unroll
  for (int s = 0; s < 2; s++) {
    qf[s].u[0] = *(const v4u*)(qptr + s * 32 + 16 * half);
    qf[s].u[1] = *(const v4u*)(qptr + s * 32 + 16 * half + 8);
  }

  v8f Oc[4];
  #pragma unroll
  for (int i = 0; i < 4; i++) Oc[i] = (v8f)0.0f;
  float mrow[8], lrow[8];
  #pragma unroll
  for (int r = 0; r < 8; r++) { mrow[r] = -3.4e38f; lrow[r] = 0.f; }

  const int ldr = tid >> 2, ldc = (tid & 3) * 16;
  const u16* kbase = kp + (long)b * NTOT * D + h * 64 + (long)ldr * D + ldc;
  const u16* vbase = vt + ((long)b * D + h * 64 + ldr) * NTOT + ldc;
  const int kvOff = ldr * 72 + ldc;
  u16* pw = Ps + wave * 16 * 72;

  auto issueKV = [&](int buf, int j0) {
    cp16(&Ks[buf][kvOff],     kbase + (long)j0 * D);
    cp16(&Ks[buf][kvOff + 8], kbase + (long)j0 * D + 8);
    cp16(&Vs[buf][kvOff],     vbase + j0);
    cp16(&Vs[buf][kvOff + 8], vbase + j0 + 8);
  };

  issueKV(0, 0);
  int cur = 0;
  for (int j0 = 0; j0 < NTOT; j0 += 64) {
    cp_waitN<0>();
    __syncthreads();
    if (j0 + 64 < NTOT) issueKV(cur ^ 1, j0 + 64);

    // S = Q @ K^T for this 64-key block (4 N-tiles, K=64 in two steps)
    v8f S[4];
    #pragma unroll
    for (int i = 0; i < 4; i++) S[i] = (v8f)0.0f;
    #pragma unroll
    for (int s = 0; s < 2; s++) {
      #pragma unroll
      for (int ni = 0; ni < 4; ni++) {
        Frag kf;
        const u16* p = &Ks[cur][(ni * 16 + l15) * 72 + s * 32 + 16 * half];
        kf.u[0] = *(const v4u*)p; kf.u[1] = *(const v4u*)(p + 8);
        S[ni] = WMMA_BF16(qf[s].b, kf.b, S[ni]);
      }
    }

    // online softmax: rows of C live within a 16-lane group -> DPP butterflies
    #pragma unroll
    for (int r = 0; r < 8; r++) {
      float mx = S[0][r];
      #pragma unroll
      for (int ni = 1; ni < 4; ni++) mx = fmaxf(mx, S[ni][r]);
      mx = grp16_max(mx);
      const float mnew = fmaxf(mrow[r], mx);
      const float alpha = __expf(mrow[r] - mnew);
      mrow[r] = mnew;
      float sum = 0.f;
      #pragma unroll
      for (int ni = 0; ni < 4; ni++) {
        const float e = __expf(S[ni][r] - mnew);
        S[ni][r] = e;
        sum += e;
      }
      sum = grp16_sum(sum);
      lrow[r] = lrow[r] * alpha + sum;
      #pragma unroll
      for (int nt = 0; nt < 4; nt++) Oc[nt][r] *= alpha;
    }

    // re-shape P: C-layout -> A-layout via wave-private LDS bounce
    #pragma unroll
    for (int ni = 0; ni < 4; ni++)
      #pragma unroll
      for (int r = 0; r < 8; r++)
        pw[(r + 8 * half) * 72 + ni * 16 + l15] = f2bf(S[ni][r]);

    // O += P @ V (K=64 keys in two steps, 4 d-tiles)
    #pragma unroll
    for (int s2 = 0; s2 < 2; s2++) {
      Frag pf;
      const u16* pp = pw + l15 * 72 + s2 * 32;
      pf.u[0] = *(const v4u*)(pp + 16 * half);
      pf.u[1] = *(const v4u*)(pp + 16 * half + 8);
      #pragma unroll
      for (int nt = 0; nt < 4; nt++) {
        Frag vf;
        const u16* vp_ = &Vs[cur][(nt * 16 + l15) * 72 + s2 * 32 + 16 * half];
        vf.u[0] = *(const v4u*)vp_; vf.u[1] = *(const v4u*)(vp_ + 8);
        Oc[nt] = WMMA_BF16(pf.b, vf.b, Oc[nt]);
      }
    }
    cur ^= 1;
  }

  const long orowBase = (long)b * NQ + mt * 128 + wave * 16 + 8 * half;
  #pragma unroll
  for (int nt = 0; nt < 4; nt++) {
    #pragma unroll
    for (int r = 0; r < 8; r++) {
      const float inv = 1.0f / lrow[r];
      ao[(orowBase + r) * D + h * 64 + nt * 16 + l15] = f2bf(Oc[nt][r] * inv);
    }
  }
}

// ---------------------------------------------------------------------------
extern "C" void kernel_launch(void* const* d_in, const int* in_sizes, int n_in,
                              void* d_out, int out_size, void* d_ws, size_t ws_size,
                              hipStream_t stream) {
  (void)in_sizes; (void)n_in; (void)out_size; (void)ws_size;
  const float* query   = (const float*)d_in[0];
  const float* kv      = (const float*)d_in[1];
  const float* ln_q_g  = (const float*)d_in[2];
  const float* ln_q_b  = (const float*)d_in[3];
  const float* ln_kv_g = (const float*)d_in[4];
  const float* ln_kv_b = (const float*)d_in[5];
  const float* Wq = (const float*)d_in[6];  const float* bq = (const float*)d_in[7];
  const float* Wk = (const float*)d_in[8];  const float* bk = (const float*)d_in[9];
  const float* Wv = (const float*)d_in[10]; const float* bv = (const float*)d_in[11];
  const float* Wo = (const float*)d_in[12]; const float* bo = (const float*)d_in[13];
  float* out = (float*)d_out;

  // workspace layout (bf16 elements), total ~76 MB
  u16* ws   = (u16*)d_ws;
  u16* WqT  = ws;
  u16* WkT  = ws + (size_t)1048576;
  u16* WvT  = ws + (size_t)2 * 1048576;
  u16* WoT  = ws + (size_t)3 * 1048576;
  u16* kvin = ws + (size_t)4 * 1048576;            // (4, 2560, 1024) ln'd concat
  u16* qp   = kvin + (size_t)4 * 2560 * 1024;      // (4*512, 1024), scaled
  u16* kp   = qp   + (size_t)4 * 512 * 1024;       // (4*2560, 1024)
  u16* vt   = kp   + (size_t)4 * 2560 * 1024;      // (4, 1024, 2560) V^T per head
  u16* ao   = vt   + (size_t)4 * 2560 * 1024;      // (4*512, 1024) attn out

  wtrans_k<<<1024, 256, 0, stream>>>(Wq, WqT);
  wtrans_k<<<1024, 256, 0, stream>>>(Wk, WkT);
  wtrans_k<<<1024, 256, 0, stream>>>(Wv, WvT);
  wtrans_k<<<1024, 256, 0, stream>>>(Wo, WoT);

  ln_bf16_k<<<1024, 256, 0, stream>>>(kv,    ln_kv_g, ln_kv_b, kvin, 2048, 0);
  ln_bf16_k<<< 256, 256, 0, stream>>>(query, ln_q_g,  ln_q_b,  kvin,  512, 2048);

  const float SCALE = 0.125f;  // 64^-0.5, folded into q projection
  gemm_bf16_k<0><<<dim3(8, 16), 256, 0, stream>>>(kvin, WqT, bq, nullptr, qp, nullptr,
      2048, 1024, 1024, 512, (long)2560 * 1024, 2048, SCALE, 0);
  gemm_bf16_k<0><<<dim3(8, 80), 256, 0, stream>>>(kvin, WkT, bk, nullptr, kp, nullptr,
      10240, 1024, 1024, 2560, (long)2560 * 1024, 0, 1.0f, 0);
  gemm_bf16_k<1><<<dim3(8, 80), 256, 0, stream>>>(kvin, WvT, bv, nullptr, vt, nullptr,
      10240, 1024, 1024, 2560, (long)2560 * 1024, 0, 1.0f, 2560);

  attn_k<<<256, 256, 0, stream>>>(qp, kp, vt, ao);

  gemm_bf16_k<2><<<dim3(8, 16), 256, 0, stream>>>(ao, WoT, bo, query, nullptr, out,
      2048, 1024, 1024, 2048, 0L, 0, 1.0f, 0);
}
